// Head_33844342292571
// MI455X (gfx1250) — matrix operand order repaired and checked by reading, not compiled, not run on hardware
//
#include <hip/hip_runtime.h>
#include <hip/hip_bf16.h>

typedef __bf16 bf16_t;
typedef __attribute__((ext_vector_type(8)))  __bf16 v8bf;
typedef __attribute__((ext_vector_type(16))) __bf16 v16bf;
typedef __attribute__((ext_vector_type(8)))  float  v8f;
typedef __attribute__((ext_vector_type(4)))  float  v4f;

#define N_EMBD 384
#define HEADSZ 64
#define BATCH  512
#define SEQ    256

__device__ __forceinline__ v16bf cat16(v8bf lo, v8bf hi) {
    return __builtin_shufflevector(lo, hi, 0,1,2,3,4,5,6,7,8,9,10,11,12,13,14,15);
}

__device__ __forceinline__ v8f wmma_bf16(v16bf a, v16bf b, v8f c) {
    // D = A(16x32 bf16) * B(32x16 bf16) + C(16x16 f32)
    return __builtin_amdgcn_wmma_f32_16x16x32_bf16(false, a, false, b, (short)0, c, false, false);
}

// A-matrix (16x32 bf16) per-lane load from a row-major bf16 matrix.
// lane<16: row=lane, K [0..15]; lane>=16: row=lane-16, K [16..31]
__device__ __forceinline__ v16bf load_a_bf16(const bf16_t* rowp /*row base for lane*/, int kbase, int hh) {
    const bf16_t* p = rowp + kbase + hh * 16;
    v8bf c0 = *(const v8bf*)(p);
    v8bf c1 = *(const v8bf*)(p + 8);
    return cat16(c0, c1);
}

// B-matrix (32x16 bf16) per-lane load: lane<16 holds K=0..15 of column lane,
// lane>=16 holds K=16..31 of column lane-16. p = data for (col, K-chunk) start.
__device__ __forceinline__ v16bf load_b_bf16(const bf16_t* p) {
    v8bf c0 = *(const v8bf*)(p);
    v8bf c1 = *(const v8bf*)(p + 8);
    return cat16(c0, c1);
}

// ---------------- kernel 0: transpose + convert weights to bf16 -------------
__global__ __launch_bounds__(256) void head_wt_kernel(
        const float* __restrict__ Wq, const float* __restrict__ Wk,
        const float* __restrict__ Wv, bf16_t* __restrict__ WT) {
    int i = blockIdx.x * 256 + threadIdx.x;              // 0 .. 3*384*64
    int m = i / (N_EMBD * HEADSZ);
    int r = i % (N_EMBD * HEADSZ);
    int c = r % HEADSZ;                                  // output column
    int k = r / HEADSZ;                                  // embedding index
    const float* W = (m == 0) ? Wq : ((m == 1) ? Wk : Wv);
    WT[(long)m * HEADSZ * N_EMBD + (long)c * N_EMBD + k] = (bf16_t)W[k * HEADSZ + c];
}

// ---------------- kernel 1: fused QKV projection (x@W + b) ------------------
// block = 256 thr = 8 waves; wave computes 16 rows x (q,k,v all 64 cols)
__global__ __launch_bounds__(256) void head_qkv_kernel(
        const float* __restrict__ x,
        const float* __restrict__ bq, const float* __restrict__ bk,
        const float* __restrict__ bv,
        const bf16_t* __restrict__ WT,
        bf16_t* __restrict__ qb, bf16_t* __restrict__ kb,
        bf16_t* __restrict__ vT) {
    const int lane = threadIdx.x & 31;
    const int wave = threadIdx.x >> 5;
    const int hrow = lane & 15;
    const int hh   = lane >> 4;
    const long row0 = (long)blockIdx.x * 128 + wave * 16;   // flat [B*T] row tile

    v8f acc[12];
    #pragma unroll
    for (int i = 0; i < 12; ++i) acc[i] = (v8f)0.0f;

    const float* xrow = x + (row0 + hrow) * N_EMBD;

    for (int kb0 = 0; kb0 < N_EMBD; kb0 += 32) {
        // A: x tile 16x32, convert f32 -> bf16 inline
        const float* p = xrow + kb0 + hh * 16;
        v4f f0 = *(const v4f*)(p);
        v4f f1 = *(const v4f*)(p + 4);
        v4f f2 = *(const v4f*)(p + 8);
        v4f f3 = *(const v4f*)(p + 12);
        v16bf a;
        #pragma unroll
        for (int i = 0; i < 4; ++i) {
            a[i]      = (__bf16)f0[i];
            a[4 + i]  = (__bf16)f1[i];
            a[8 + i]  = (__bf16)f2[i];
            a[12 + i] = (__bf16)f3[i];
        }
        #pragma unroll
        for (int m = 0; m < 3; ++m) {
            const bf16_t* Wm = WT + (long)m * HEADSZ * N_EMBD;
            #pragma unroll
            for (int nt = 0; nt < 4; ++nt) {
                // B: W^T column (nt*16 + hrow), K slice contiguous
                const bf16_t* bp = Wm + (long)(nt * 16 + hrow) * N_EMBD + kb0 + hh * 16;
                acc[m * 4 + nt] = wmma_bf16(a, load_b_bf16(bp), acc[m * 4 + nt]);
            }
        }
    }

    const long bidx = row0 / SEQ;
    const long t0   = row0 % SEQ;

    #pragma unroll
    for (int m = 0; m < 3; ++m) {
        const float* bias = (m == 0) ? bq : ((m == 1) ? bk : bv);
        #pragma unroll
        for (int nt = 0; nt < 4; ++nt) {
            float bcol = bias[nt * 16 + hrow];
            v8f d = acc[m * 4 + nt];
            if (m < 2) {
                // q,k: row-major bf16 [B*T, 64]
                bf16_t* o = ((m == 0) ? qb : kb);
                #pragma unroll
                for (int r = 0; r < 8; ++r) {
                    long row = row0 + hh * 8 + r;           // D row = r + hh*8
                    o[row * HEADSZ + nt * 16 + hrow] = (bf16_t)(d[r] + bcol);
                }
            } else {
                // v: transposed bf16 [B, 64, T]; 8 consecutive t per lane -> packed b128
                v8bf pk;
                #pragma unroll
                for (int r = 0; r < 8; ++r) pk[r] = (bf16_t)(d[r] + bcol);
                bf16_t* o = vT + bidx * (HEADSZ * SEQ)
                               + (long)(nt * 16 + hrow) * SEQ + t0 + hh * 8;
                *(v8bf*)o = pk;
            }
        }
    }
}

// ---------------- kernel 2: flash attention over one 16-query tile per wave -
__global__ __launch_bounds__(256) void head_attn_kernel(
        const bf16_t* __restrict__ qb, const bf16_t* __restrict__ kb,
        const bf16_t* __restrict__ vT, float* __restrict__ out) {
    __shared__ __align__(16) bf16_t Pl[8][16 * 32];   // per-wave P scratch (D->A relayout)

    const int lane = threadIdx.x & 31;
    const int wave = threadIdx.x >> 5;
    const int hrow = lane & 15;
    const int hh   = lane >> 4;

    const long rows128 = (long)blockIdx.x * 128;
    const long bidx = rows128 / SEQ;
    const int  t0   = (int)(rows128 % SEQ) + wave * 16;

    const float scale = 0.05103103630798288f;         // 384^-0.5 (ref scales by n_embd)

    const bf16_t* kbase = kb + bidx * SEQ * HEADSZ;
    const bf16_t* vbase = vT + bidx * HEADSZ * SEQ;
    bf16_t* myP = &Pl[wave][0];

    // q tile as A-matrix, two K-steps over head dim (resident for whole kernel)
    const bf16_t* qrow = qb + (bidx * SEQ + t0 + hrow) * HEADSZ;
    v16bf a0 = load_a_bf16(qrow, 0, hh);
    v16bf a1 = load_a_bf16(qrow, 32, hh);

    v8f o[4];
    #pragma unroll
    for (int nt = 0; nt < 4; ++nt) o[nt] = (v8f)0.0f;
    float mrow[8], lrow[8];
    #pragma unroll
    for (int r = 0; r < 8; ++r) { mrow[r] = -1e30f; lrow[r] = 0.0f; }

    const int ngroups = (t0 + 15) / 32 + 1;           // causal bound, 32-key groups

    for (int g = 0; g < ngroups; ++g) {
        const int kb0 = g * 32;
        if (g + 1 < ngroups)
            __builtin_prefetch(kbase + (long)(kb0 + 32) * HEADSZ, 0, 1);

        // S = q @ k^T : two 16x16 f32 tiles (keys kb0.. and kb0+16..)
        v8f s0 = (v8f)0.0f, s1 = (v8f)0.0f;
        {
            const bf16_t* b00 = kbase + (long)(kb0 + hrow) * HEADSZ + 0  + hh * 16;
            const bf16_t* b01 = kbase + (long)(kb0 + hrow) * HEADSZ + 32 + hh * 16;
            s0 = wmma_bf16(a0, load_b_bf16(b00), s0);
            s0 = wmma_bf16(a1, load_b_bf16(b01), s0);
            const bf16_t* b10 = kbase + (long)(kb0 + 16 + hrow) * HEADSZ + 0  + hh * 16;
            const bf16_t* b11 = kbase + (long)(kb0 + 16 + hrow) * HEADSZ + 32 + hh * 16;
            s1 = wmma_bf16(a0, load_b_bf16(b10), s1);
            s1 = wmma_bf16(a1, load_b_bf16(b11), s1);
        }

        // online softmax update (rows live across 16 lanes of each half-wave)
        const int kj0 = kb0 + hrow;
        const int kj1 = kb0 + 16 + hrow;
        float p0[8], p1[8];
        #pragma unroll
        for (int r = 0; r < 8; ++r) {
            const int qr = t0 + hh * 8 + r;
            float x0 = (kj0 <= qr) ? s0[r] * scale : -1e30f;
            float x1 = (kj1 <= qr) ? s1[r] * scale : -1e30f;
            float mx = fmaxf(x0, x1);
            #pragma unroll
            for (int off = 1; off < 16; off <<= 1)
                mx = fmaxf(mx, __shfl_xor(mx, off, 32));
            const float mnew = fmaxf(mrow[r], mx);
            const float corr = __expf(mrow[r] - mnew);
            const float e0 = __expf(x0 - mnew);
            const float e1 = __expf(x1 - mnew);
            float rs = e0 + e1;
            #pragma unroll
            for (int off = 1; off < 16; off <<= 1)
                rs += __shfl_xor(rs, off, 32);
            lrow[r] = lrow[r] * corr + rs;
            mrow[r] = mnew;
            p0[r] = e0; p1[r] = e1;
            #pragma unroll
            for (int nt = 0; nt < 4; ++nt) o[nt][r] = o[nt][r] * corr;
        }

        // D-layout P -> row-major LDS tile [16 x 32] (wave-private, DS in-order)
        #pragma unroll
        for (int r = 0; r < 8; ++r) {
            const int m = r + hh * 8;
            myP[m * 32 + hrow]      = (bf16_t)p0[r];
            myP[m * 32 + 16 + hrow] = (bf16_t)p1[r];
        }
        // reload as A-matrix (16x32 bf16)
        const bf16_t* pp = myP + hrow * 32 + hh * 16;
        v8bf pc0 = *(const v8bf*)(pp);
        v8bf pc1 = *(const v8bf*)(pp + 8);
        v16bf pa = cat16(pc0, pc1);

        // O += P @ V : B from transposed V (contiguous per-lane key slices)
        #pragma unroll
        for (int nt = 0; nt < 4; ++nt) {
            const bf16_t* bp = vbase + (long)(nt * 16 + hrow) * SEQ + kb0 + hh * 16;
            o[nt] = wmma_bf16(pa, load_b_bf16(bp), o[nt]);
        }
    }

    // finalize: divide by row sums, store f32 [B, T, 64]
    float* obase = out + bidx * SEQ * (long)HEADSZ;
    #pragma unroll
    for (int r = 0; r < 8; ++r) {
        const float inv = 1.0f / lrow[r];
        const int trow = t0 + hh * 8 + r;
        #pragma unroll
        for (int nt = 0; nt < 4; ++nt)
            obase[(long)trow * HEADSZ + nt * 16 + hrow] = o[nt][r] * inv;
    }
}

extern "C" void kernel_launch(void* const* d_in, const int* in_sizes, int n_in,
                              void* d_out, int out_size, void* d_ws, size_t ws_size,
                              hipStream_t stream) {
    const float* x  = (const float*)d_in[0];
    const float* Wq = (const float*)d_in[1];
    const float* bq = (const float*)d_in[2];
    const float* Wk = (const float*)d_in[3];
    const float* bk = (const float*)d_in[4];
    const float* Wv = (const float*)d_in[5];
    const float* bv = (const float*)d_in[6];
    float* out = (float*)d_out;

    char* ws = (char*)d_ws;
    const size_t wt_bytes  = (size_t)3 * HEADSZ * N_EMBD * sizeof(bf16_t);   // 147456
    const size_t mat_bytes = (size_t)BATCH * SEQ * HEADSZ * sizeof(bf16_t);  // 16 MB each
    bf16_t* WT = (bf16_t*)(ws);
    bf16_t* qb = (bf16_t*)(ws + wt_bytes);
    bf16_t* kb = (bf16_t*)(ws + wt_bytes + mat_bytes);
    bf16_t* vT = (bf16_t*)(ws + wt_bytes + 2 * mat_bytes);

    // 1) weights -> bf16 transposed
    head_wt_kernel<<<(3 * N_EMBD * HEADSZ) / 256, 256, 0, stream>>>(Wq, Wk, Wv, WT);
    // 2) fused QKV projection: 131072 rows / 128 rows per block
    head_qkv_kernel<<<(BATCH * SEQ) / 128, 256, 0, stream>>>(x, bq, bk, bv, WT, qb, kb, vT);
    // 3) flash attention
    head_attn_kernel<<<(BATCH * SEQ) / 128, 256, 0, stream>>>(qb, kb, vT, out);
}
